// MultiHeadAttention_3805341024481
// MI455X (gfx1250) — compile-verified, matrix operand order checked by reading, not run on hardware
//
#include <hip/hip_runtime.h>

#define D_MODEL 1024
#define NUM_HEADS 16
#define HEAD_DIM 64
#define BATCH 2
#define SEQ 2048
#define MTOT (BATCH * SEQ)   // 4096 rows

typedef __bf16 bf16;
typedef __attribute__((ext_vector_type(16))) __bf16 v16bf;
typedef __attribute__((ext_vector_type(8)))  __bf16 v8bf;
typedef __attribute__((ext_vector_type(4)))  __bf16 v4bf;
typedef __attribute__((ext_vector_type(8)))  float  v8f;
typedef __attribute__((ext_vector_type(4)))  float  v4f;
typedef __attribute__((ext_vector_type(4)))  int    v4i;

#if __has_builtin(__builtin_amdgcn_global_load_async_to_lds_b128)
#define USE_ASYNC_LDS 1
typedef __attribute__((address_space(1))) v4i* gv4i_p;  // global int4*
typedef __attribute__((address_space(3))) v4i* lv4i_p;  // LDS int4*
#else
#define USE_ASYNC_LDS 0
#endif

static __device__ __forceinline__ void wait_async0() {
#if __has_builtin(__builtin_amdgcn_s_wait_asynccnt)
  __builtin_amdgcn_s_wait_asynccnt(0);
#else
  asm volatile("s_wait_asynccnt 0" ::: "memory");
#endif
}

static __device__ __forceinline__ v8f zero8() {
  v8f z;
#pragma unroll
  for (int i = 0; i < 8; ++i) z[i] = 0.0f;
  return z;
}

// A-fragment / B-fragment loader (identical lane layout for A and B^T):
// lane L: row = L&15, two contiguous 8-element chunks at k0=(L>>4)*8 and k0+16.
static __device__ __forceinline__ v16bf load_frag(const bf16* __restrict__ base,
                                                  int rowStride, int lane) {
  const int r  = lane & 15;
  const int k0 = (lane >> 4) << 3;
  const bf16* p = base + (size_t)r * rowStride + k0;
  v8bf lo = *(const v8bf*)(p);
  v8bf hi = *(const v8bf*)(p + 16);
  return __builtin_shufflevector(lo, hi, 0, 1, 2, 3, 4, 5, 6, 7,
                                 8, 9, 10, 11, 12, 13, 14, 15);
}

static __device__ __forceinline__ v8f wmma_bf16(v16bf a, v16bf b, v8f c) {
  return __builtin_amdgcn_wmma_f32_16x16x32_bf16(false, a, false, b, (short)0, c,
                                                 false, false);
}

// ---------------------------------------------------------------------------
// Kernel 1: fused Q/K/V projection.  C[m,n] = X[m,:] . W[n,:] + bias[n]
// Block tile 64M x 128N, 4 waves (2x2), each wave 32M x 64N (2x4 WMMA tiles).
// blockIdx.z selects {Q, K, V}.  Q scaled by 1/sqrt(HEAD_DIM), stored [B,H,S,Dh];
// K stored [B,H,S,Dh]; V stored transposed [B,H,Dh,S].
// ---------------------------------------------------------------------------
#define LDSK 40  // padded LDS row stride (bf16 elems) -> conflict-free 16B reads

__global__ __launch_bounds__(128) void qkv_proj_kernel(
    const float* __restrict__ Xq, const float* __restrict__ Xk,
    const float* __restrict__ Xv, const float* __restrict__ Wq,
    const float* __restrict__ Wk, const float* __restrict__ Wv,
    const float* __restrict__ bq, const float* __restrict__ bk,
    const float* __restrict__ bv, bf16* __restrict__ Qh,
    bf16* __restrict__ Kh, bf16* __restrict__ Vt) {
  __shared__ bf16 As[64 * LDSK];
  __shared__ bf16 Bs[128 * LDSK];

  const int which = blockIdx.z;  // 0=Q 1=K 2=V
  const float* X    = (which == 0) ? Xq : (which == 1) ? Xk : Xv;
  const float* W    = (which == 0) ? Wq : (which == 1) ? Wk : Wv;
  const float* bias = (which == 0) ? bq : (which == 1) ? bk : bv;

  const int m0   = blockIdx.y * 64;
  const int n0   = blockIdx.x * 128;
  const int tid  = threadIdx.x;
  const int lane = tid & 31;
  const int wave = tid >> 5;
  const int wm   = (wave >> 1) * 32;
  const int wn   = (wave & 1) * 64;

  v8f acc[2][4];
#pragma unroll
  for (int i = 0; i < 2; ++i)
#pragma unroll
    for (int j = 0; j < 4; ++j) acc[i][j] = zero8();

  for (int kk = 0; kk < D_MODEL; kk += 32) {
    // Stage A (X, 64x32) and B (W, 128x32) tiles, f32 -> bf16, into LDS.
#pragma unroll
    for (int it = 0; it < 4; ++it) {
      const int idx = (it * 128 + tid) * 4;  // 0..2047
      const int r = idx >> 5, c = idx & 31;
      v4f a = *(const v4f*)(X + (size_t)(m0 + r) * D_MODEL + kk + c);
      v4bf ab;
#pragma unroll
      for (int e = 0; e < 4; ++e) ab[e] = (bf16)a[e];
      *(v4bf*)(&As[r * LDSK + c]) = ab;
    }
#pragma unroll
    for (int it = 0; it < 8; ++it) {
      const int idx = (it * 128 + tid) * 4;  // 0..4095
      const int r = idx >> 5, c = idx & 31;
      v4f b = *(const v4f*)(W + (size_t)(n0 + r) * D_MODEL + kk + c);
      v4bf bb;
#pragma unroll
      for (int e = 0; e < 4; ++e) bb[e] = (bf16)b[e];
      *(v4bf*)(&Bs[r * LDSK + c]) = bb;
    }
    __syncthreads();

    v16bf a0 = load_frag(&As[(wm + 0)  * LDSK], LDSK, lane);
    v16bf a1 = load_frag(&As[(wm + 16) * LDSK], LDSK, lane);
#pragma unroll
    for (int t = 0; t < 4; ++t) {
      v16bf bt = load_frag(&Bs[(wn + t * 16) * LDSK], LDSK, lane);
      acc[0][t] = wmma_bf16(a0, bt, acc[0][t]);
      acc[1][t] = wmma_bf16(a1, bt, acc[1][t]);
    }
    __syncthreads();
  }

  const int half = lane >> 4, colp = lane & 15;
#pragma unroll
  for (int mi = 0; mi < 2; ++mi)
#pragma unroll
    for (int t = 0; t < 4; ++t)
#pragma unroll
      for (int j = 0; j < 8; ++j) {
        const int m = m0 + wm + mi * 16 + j + half * 8;
        const int n = n0 + wn + t * 16 + colp;
        const float v = acc[mi][t][j] + bias[n];
        const int bb = m >> 11, s = m & (SEQ - 1);
        const int hh = n >> 6, d = n & (HEAD_DIM - 1);
        if (which == 0)
          Qh[(((size_t)bb * NUM_HEADS + hh) * SEQ + s) * HEAD_DIM + d] =
              (bf16)(v * 0.125f);  // fold 1/sqrt(64)
        else if (which == 1)
          Kh[(((size_t)bb * NUM_HEADS + hh) * SEQ + s) * HEAD_DIM + d] = (bf16)v;
        else
          Vt[(((size_t)bb * NUM_HEADS + hh) * HEAD_DIM + d) * SEQ + s] = (bf16)v;
      }
}

// ---------------------------------------------------------------------------
// Kernel 2: flash attention.  One block per (b*h, 64-row q tile).
// Each wave owns 16 q rows; streams 64-wide K/V tiles with online softmax.
// ---------------------------------------------------------------------------
#define PST 72  // padded LDS stride for P re-layout buffer

__global__ __launch_bounds__(128) void flash_attn_kernel(
    const bf16* __restrict__ Qh, const bf16* __restrict__ Kh,
    const bf16* __restrict__ Vt, bf16* __restrict__ AO) {
  __shared__ bf16 Pbuf[4][16 * PST];

  const int qt = blockIdx.x;        // q tile (64 rows)
  const int bh = blockIdx.y;        // b*NUM_HEADS + h
  const int b = bh >> 4, h = bh & 15;
  const int tid = threadIdx.x, lane = tid & 31, wave = tid >> 5;
  const int q0 = qt * 64 + wave * 16;

  const bf16* Qbase = Qh + (size_t)bh * SEQ * HEAD_DIM;
  const bf16* Kbase = Kh + (size_t)bh * SEQ * HEAD_DIM;
  const bf16* Vbase = Vt + (size_t)bh * HEAD_DIM * SEQ;

  // Q A-fragments for dh in {0..31} and {32..63}: kept resident all kernel.
  v16bf qf0 = load_frag(Qbase + (size_t)q0 * HEAD_DIM + 0, HEAD_DIM, lane);
  v16bf qf1 = load_frag(Qbase + (size_t)q0 * HEAD_DIM + 32, HEAD_DIM, lane);

  v8f o[4];
#pragma unroll
  for (int nt = 0; nt < 4; ++nt) o[nt] = zero8();
  float m_[8], l_[8];
#pragma unroll
  for (int j = 0; j < 8; ++j) { m_[j] = -3.0e38f; l_[j] = 0.0f; }

  bf16* pb = &Pbuf[wave][0];
  const int half = lane >> 4, colp = lane & 15;

  for (int kt = 0; kt < SEQ; kt += 64) {
    // Prefetch next K tile rows into cache (global_prefetch_b8 path).
    if (kt + 64 < SEQ)
      __builtin_prefetch(Kbase + (size_t)(kt + 64 + (lane & 63)) * HEAD_DIM, 0, 1);

    // ---- scores S = Q . K^T for 16 q-rows x 64 k-cols (4 WMMA n-tiles) ----
    v8f st[4];
#pragma unroll
    for (int nt = 0; nt < 4; ++nt) {
      const bf16* kb = Kbase + (size_t)(kt + nt * 16) * HEAD_DIM;
      v16bf kb0 = load_frag(kb + 0, HEAD_DIM, lane);
      v16bf kb1 = load_frag(kb + 32, HEAD_DIM, lane);
      v8f s = zero8();
      s = wmma_bf16(qf0, kb0, s);
      s = wmma_bf16(qf1, kb1, s);
      st[nt] = s;
    }

    // ---- online softmax (row = j + 8*half; xor masks stay in 16-lane half)
#pragma unroll
    for (int j = 0; j < 8; ++j) {
      float rm = fmaxf(fmaxf(st[0][j], st[1][j]), fmaxf(st[2][j], st[3][j]));
      rm = fmaxf(rm, __shfl_xor(rm, 1));
      rm = fmaxf(rm, __shfl_xor(rm, 2));
      rm = fmaxf(rm, __shfl_xor(rm, 4));
      rm = fmaxf(rm, __shfl_xor(rm, 8));
      const float mnew = fmaxf(m_[j], rm);
      const float corr = __expf(m_[j] - mnew);
      float rs = 0.0f;
#pragma unroll
      for (int nt = 0; nt < 4; ++nt) {
        const float p = __expf(st[nt][j] - mnew);
        st[nt][j] = p;
        rs += p;
      }
      rs += __shfl_xor(rs, 1);
      rs += __shfl_xor(rs, 2);
      rs += __shfl_xor(rs, 4);
      rs += __shfl_xor(rs, 8);
      l_[j] = l_[j] * corr + rs;
      m_[j] = mnew;
#pragma unroll
      for (int nt = 0; nt < 4; ++nt) o[nt][j] *= corr;
    }

    // ---- re-layout P (C-layout -> A-fragment layout) through LDS ----
#pragma unroll
    for (int nt = 0; nt < 4; ++nt)
#pragma unroll
      for (int j = 0; j < 8; ++j)
        pb[(j + half * 8) * PST + nt * 16 + colp] = (bf16)st[nt][j];
    __syncthreads();

    // ---- O += P . V (two 32-deep k-steps) ----
#pragma unroll
    for (int ks = 0; ks < 64; ks += 32) {
      v16bf pa = load_frag(pb + ks, PST, lane);
#pragma unroll
      for (int nt = 0; nt < 4; ++nt) {
        const bf16* vb = Vbase + (size_t)(nt * 16) * SEQ + kt + ks;
        v16bf vf = load_frag(vb, SEQ, lane);
        o[nt] = wmma_bf16(pa, vf, o[nt]);
      }
    }
    __syncthreads();
  }

  // ---- normalize and emit attention output bf16 [M, D_MODEL] ----
#pragma unroll
  for (int j = 0; j < 8; ++j) {
    const float inv = 1.0f / l_[j];
    const int q = q0 + j + half * 8;
    const size_t row = (size_t)b * SEQ + q;
#pragma unroll
    for (int nt = 0; nt < 4; ++nt) {
      const int d = h * HEAD_DIM + nt * 16 + colp;
      AO[row * D_MODEL + d] = (bf16)(o[nt][j] * inv);
    }
  }
}

// ---------------------------------------------------------------------------
// Kernel 3: output projection.  out[m,n] = AO[m,:].Wo[n,:] + bo[n]  (fp32 out)
// Block tile 64M x 128N.  A tile staged via async global->LDS when available.
// ---------------------------------------------------------------------------
__global__ __launch_bounds__(128) void out_proj_kernel(
    const bf16* __restrict__ AO, const float* __restrict__ Wo,
    const float* __restrict__ bo, float* __restrict__ out) {
  __shared__ bf16 As[64 * LDSK];
  __shared__ bf16 Bs[128 * LDSK];

  const int m0   = blockIdx.y * 64;
  const int n0   = blockIdx.x * 128;
  const int tid  = threadIdx.x;
  const int lane = tid & 31;
  const int wave = tid >> 5;
  const int wm   = (wave >> 1) * 32;
  const int wn   = (wave & 1) * 64;

  v8f acc[2][4];
#pragma unroll
  for (int i = 0; i < 2; ++i)
#pragma unroll
    for (int j = 0; j < 4; ++j) acc[i][j] = zero8();

  for (int kk = 0; kk < D_MODEL; kk += 32) {
    // A tile already bf16: async DMA straight into LDS (no VGPR round trip).
#pragma unroll
    for (int it = 0; it < 2; ++it) {
      const int idx = (it * 128 + tid) * 8;  // 0..2047
      const int r = idx >> 5, c = idx & 31;
      const bf16* src = AO + (size_t)(m0 + r) * D_MODEL + kk + c;
      bf16* dst = &As[r * LDSK + c];
#if USE_ASYNC_LDS
      __builtin_amdgcn_global_load_async_to_lds_b128((gv4i_p)(src), (lv4i_p)(dst),
                                                     0, 0);
#else
      *(v8bf*)dst = *(const v8bf*)src;
#endif
    }
    // B tile: Wo f32 -> bf16 through VGPRs (conversion needed anyway).
#pragma unroll
    for (int it = 0; it < 8; ++it) {
      const int idx = (it * 128 + tid) * 4;  // 0..4095
      const int r = idx >> 5, c = idx & 31;
      v4f w = *(const v4f*)(Wo + (size_t)(n0 + r) * D_MODEL + kk + c);
      v4bf wb;
#pragma unroll
      for (int e = 0; e < 4; ++e) wb[e] = (bf16)w[e];
      *(v4bf*)(&Bs[r * LDSK + c]) = wb;
    }
#if USE_ASYNC_LDS
    wait_async0();
#endif
    __syncthreads();

    v16bf a0 = load_frag(&As[(wm + 0)  * LDSK], LDSK, lane);
    v16bf a1 = load_frag(&As[(wm + 16) * LDSK], LDSK, lane);
#pragma unroll
    for (int t = 0; t < 4; ++t) {
      v16bf bt = load_frag(&Bs[(wn + t * 16) * LDSK], LDSK, lane);
      acc[0][t] = wmma_bf16(a0, bt, acc[0][t]);
      acc[1][t] = wmma_bf16(a1, bt, acc[1][t]);
    }
    __syncthreads();
  }

  const int half = lane >> 4, colp = lane & 15;
#pragma unroll
  for (int mi = 0; mi < 2; ++mi)
#pragma unroll
    for (int t = 0; t < 4; ++t)
#pragma unroll
      for (int j = 0; j < 8; ++j) {
        const int m = m0 + wm + mi * 16 + j + half * 8;
        const int n = n0 + wn + t * 16 + colp;
        out[(size_t)m * D_MODEL + n] = acc[mi][t][j] + bo[n];
      }
}

// ---------------------------------------------------------------------------
extern "C" void kernel_launch(void* const* d_in, const int* in_sizes, int n_in,
                              void* d_out, int out_size, void* d_ws,
                              size_t ws_size, hipStream_t stream) {
  const float* Xq = (const float*)d_in[0];
  const float* Xk = (const float*)d_in[1];
  const float* Xv = (const float*)d_in[2];
  const float* Wq = (const float*)d_in[3];
  const float* bq = (const float*)d_in[4];
  const float* Wk = (const float*)d_in[5];
  const float* bk = (const float*)d_in[6];
  const float* Wv = (const float*)d_in[7];
  const float* bv = (const float*)d_in[8];
  const float* Wo = (const float*)d_in[9];
  const float* bo = (const float*)d_in[10];
  float* out = (float*)d_out;

  const size_t NE = (size_t)MTOT * D_MODEL;  // 4M elems per tensor
  bf16* Qh = (bf16*)d_ws;
  bf16* Kh = Qh + NE;
  bf16* Vt = Kh + NE;
  bf16* AO = Vt + NE;

  dim3 blk(128);
  dim3 grid_proj(D_MODEL / 128, MTOT / 64, 3);  // 8 x 64 x {Q,K,V}
  qkv_proj_kernel<<<grid_proj, blk, 0, stream>>>(Xq, Xk, Xv, Wq, Wk, Wv, bq, bk,
                                                 bv, Qh, Kh, Vt);

  dim3 grid_fa(SEQ / 64, BATCH * NUM_HEADS);  // 32 x 32
  flash_attn_kernel<<<grid_fa, blk, 0, stream>>>(Qh, Kh, Vt, AO);

  dim3 grid_out(D_MODEL / 128, MTOT / 64);  // 8 x 64
  out_proj_kernel<<<grid_out, blk, 0, stream>>>(AO, Wo, bo, out);
}
